// EdgeDecoder_5153960755633
// MI455X (gfx1250) — matrix operand order repaired and checked by reading, not compiled, not run on hardware
//
#include <hip/hip_runtime.h>

// ---------------------------------------------------------------------------
// EdgeDecoder: out[e] = w2 . relu( concat(z_sotu[row[e]], z_taxon[col[e]]) @ W1^T + b1 ) + b2
// H = 128, 2H = 256.  WMMA f16 path with f32 accumulation (CDNA5 / gfx1250).
// - double-buffered LDS activation tile hides random gather behind WMMA chain
// - padded LDS stride (264 f16) -> conflict-free ds_load_b128 A fragments
// - sched_group_barrier directives pipeline A-frag DS reads against WMMAs
// ---------------------------------------------------------------------------

typedef _Float16 v8h  __attribute__((ext_vector_type(8)));
typedef _Float16 v16h __attribute__((ext_vector_type(16)));
typedef float    v4f  __attribute__((ext_vector_type(4)));
typedef float    v8f  __attribute__((ext_vector_type(8)));

#define HDIM     128
#define TWO_H    256
#define ZSTRIDE  264         // 256 + 8 f16 pad: 528 B row = 132 dwords = 4 (mod 64 banks)
#define M_TILE   16          // edges per WMMA tile
#define TILES_PB 8           // M-tiles per block (amortizes w1 fragment preload)
#define EPB      (M_TILE * TILES_PB)
#define NWAVES   8

#if defined(__has_builtin)
#  if __has_builtin(__builtin_amdgcn_sched_group_barrier)
#    define SGB(mask, cnt, id) __builtin_amdgcn_sched_group_barrier((mask), (cnt), (id))
#  endif
#endif
#ifndef SGB
#  define SGB(mask, cnt, id)
#endif

__global__ __launch_bounds__(256)
void edge_decoder_wmma(const float* __restrict__ z_sotu,
                       const float* __restrict__ z_taxon,
                       const int*   __restrict__ row,
                       const int*   __restrict__ col,
                       const float* __restrict__ w1,    // [128][256]
                       const float* __restrict__ b1,    // [128]
                       const float* __restrict__ w2,    // [1][128]
                       const float* __restrict__ b2,    // [1]
                       float* __restrict__ out,         // [n_edges]
                       int n_edges)
{
    __shared__ __align__(16) _Float16 ztile[2][M_TILE * ZSTRIDE]; // 2 x 8.25 KB
    __shared__ float partial[2][NWAVES][M_TILE];                  // per-wave row partials

    const int tid  = threadIdx.x;
    const int wave = tid >> 5;              // 0..7 : output-feature slice
    const int lane = tid & 31;
    const int nloc = lane & 15;             // B/C column within tile; also A-row id
    const int ncol = (wave << 4) | nloc;    // global output feature 0..127
    const int hiK  = (lane >> 4) & 1;       // which K-half of each 32-chunk

    // ---- Preload this wave's B fragments: w1[ncol][k], f32 -> f16, all 8 K-chunks.
    // B (32x16 f16) layout: lanes 0-15 hold K=0..15 of the chunk, lanes 16-31 K=16..31.
    v16h Bfrag[8];
    {
        const float* wrow = w1 + (size_t)ncol * TWO_H + hiK * 16;
        #pragma unroll
        for (int c = 0; c < 8; ++c) {
            v16h b;
            #pragma unroll
            for (int q = 0; q < 4; ++q) {
                v4f v = *(const v4f*)(wrow + c * 32 + q * 4);
                b[q * 4 + 0] = (_Float16)v[0];
                b[q * 4 + 1] = (_Float16)v[1];
                b[q * 4 + 2] = (_Float16)v[2];
                b[q * 4 + 3] = (_Float16)v[3];
            }
            Bfrag[c] = b;
        }
    }
    const float b1n = b1[ncol];
    const float w2n = w2[ncol];
    const float b2v = b2[0];

    // ---- gather-role assignment (fixed for this thread across all tiles)
    const int e0 = blockIdx.x * EPB;
    const int el = tid & 15;                 // edge within tile
    const int fc = tid >> 4;                 // 16-float feature chunk, 0..15
    const float* ztab = (fc < 8) ? z_sotu : z_taxon;
    const int    fo   = ((fc < 8) ? fc : (fc - 8)) * 16;

    // preload all tile gather indices (breaks index->embedding dependent chain)
    int idx[TILES_PB];
    #pragma unroll
    for (int t = 0; t < TILES_PB; ++t) {
        const int e = min(e0 + t * M_TILE + el, n_edges - 1);
        idx[t] = (fc < 8) ? row[e] : col[e];
    }

    // ---- prologue: gather + convert tile 0 into LDS buffer 0
    {
        const float* src = ztab + (size_t)idx[0] * HDIM + fo;
        _Float16* dst = &ztile[0][el * ZSTRIDE + fc * 16];
        #pragma unroll
        for (int q = 0; q < 4; ++q) {
            v4f v = *(const v4f*)(src + q * 4);
            dst[q * 4 + 0] = (_Float16)v[0];
            dst[q * 4 + 1] = (_Float16)v[1];
            dst[q * 4 + 2] = (_Float16)v[2];
            dst[q * 4 + 3] = (_Float16)v[3];
        }
    }
    __syncthreads();

    for (int t = 0; t < TILES_PB; ++t) {
        const int buf = t & 1;

        // ---- issue NEXT tile's global gathers now; waits land after the WMMAs
        v4f g[4];
        if (t + 1 < TILES_PB) {
            const float* src = ztab + (size_t)idx[t + 1] * HDIM + fo;
            #pragma unroll
            for (int q = 0; q < 4; ++q) g[q] = *(const v4f*)(src + q * 4);
        }

        // ---- A fragments from LDS, WMMA chain.
        // A (16x32 f16) layout: lane<16 owns K 0..15, lane>=16 owns K 16..31.
        const _Float16* arow = &ztile[buf][nloc * ZSTRIDE + hiK * 16];
        v16h Afrag[8];
        #pragma unroll
        for (int c = 0; c < 8; ++c) {
            v8h lo = *(const v8h*)(arow + c * 32);        // K = 32c + off .. +8
            v8h hi = *(const v8h*)(arow + c * 32 + 8);    // K = 32c + off + 8 .. +8
            Afrag[c] = __builtin_shufflevector(lo, hi,
                         0, 1, 2, 3, 4, 5, 6, 7, 8, 9, 10, 11, 12, 13, 14, 15);
        }
        v8f C = {0.f, 0.f, 0.f, 0.f, 0.f, 0.f, 0.f, 0.f};
        #pragma unroll
        for (int c = 0; c < 8; ++c) {
            C = __builtin_amdgcn_wmma_f32_16x16x32_f16(
                    /*neg_a=*/false, Afrag[c], /*neg_b=*/false, Bfrag[c],
                    /*c_mod=*/(short)0, C, /*reuse_a=*/false, /*reuse_b=*/false);
        }

        // ---- scheduling directives for this region (best effort):
        // gathers first, then a depth-2 pipeline of DS-reads vs WMMAs:
        //   DS x4, [WMMA x1, DS x2] x6, WMMA x2
        SGB(0x020, 4, 0);            // 4x global_load_b128 (next tile gather)
        SGB(0x100, 4, 0);            // A frags 0,1
        SGB(0x008, 1, 0); SGB(0x100, 2, 0);   // wmma0  | loads frag2
        SGB(0x008, 1, 0); SGB(0x100, 2, 0);   // wmma1  | loads frag3
        SGB(0x008, 1, 0); SGB(0x100, 2, 0);   // wmma2  | loads frag4
        SGB(0x008, 1, 0); SGB(0x100, 2, 0);   // wmma3  | loads frag5
        SGB(0x008, 1, 0); SGB(0x100, 2, 0);   // wmma4  | loads frag6
        SGB(0x008, 1, 0); SGB(0x100, 2, 0);   // wmma5  | loads frag7
        SGB(0x008, 2, 0);                     // wmma6, wmma7

        // ---- epilogue: h = relu(C + b1[n]); reduce h*w2[n] over the 16 n-lanes.
        // C layout: lane<16 -> rows 0..7 in C[0..7]; lane>=16 -> rows 8..15.
        float s[8];
        #pragma unroll
        for (int j = 0; j < 8; ++j) {
            float h = C[j] + b1n;
            h = fmaxf(h, 0.0f);
            float v = h * w2n;
            v += __shfl_xor(v, 1, 32);
            v += __shfl_xor(v, 2, 32);
            v += __shfl_xor(v, 4, 32);
            v += __shfl_xor(v, 8, 32);
            s[j] = v;
        }
        if (nloc == 0) {                       // lanes 0 and 16 only: one branch
            float* p = &partial[buf][wave][hiK * 8];
            #pragma unroll
            for (int j = 0; j < 8; ++j) p[j] = s[j];
        }

        // ---- convert + store NEXT tile into the other LDS buffer
        if (t + 1 < TILES_PB) {
            _Float16* dst = &ztile[buf ^ 1][el * ZSTRIDE + fc * 16];
            #pragma unroll
            for (int q = 0; q < 4; ++q) {
                dst[q * 4 + 0] = (_Float16)g[q][0];
                dst[q * 4 + 1] = (_Float16)g[q][1];
                dst[q * 4 + 2] = (_Float16)g[q][2];
                dst[q * 4 + 3] = (_Float16)g[q][3];
            }
        }

        __syncthreads();   // single barrier per tile: partials done + next buf ready

        if (tid < M_TILE) {
            const int e = e0 + t * M_TILE + tid;
            if (e < n_edges) {
                float a = 0.0f;
                #pragma unroll
                for (int w = 0; w < NWAVES; ++w) a += partial[buf][w][tid];
                out[e] = a + b2v;
            }
        }
    }
}

// ---------------------------------------------------------------------------
// Harness entry point.
// d_in order: z_sotu, z_taxon, row, col, w1, b1, w2, b2
// ---------------------------------------------------------------------------
extern "C" void kernel_launch(void* const* d_in, const int* in_sizes, int n_in,
                              void* d_out, int out_size, void* d_ws, size_t ws_size,
                              hipStream_t stream) {
    const float* z_sotu  = (const float*)d_in[0];
    const float* z_taxon = (const float*)d_in[1];
    const int*   row     = (const int*)  d_in[2];
    const int*   col     = (const int*)  d_in[3];
    const float* w1      = (const float*)d_in[4];
    const float* b1      = (const float*)d_in[5];
    const float* w2      = (const float*)d_in[6];
    const float* b2      = (const float*)d_in[7];
    float* out = (float*)d_out;

    const int n_edges = in_sizes[2];           // 1,000,000
    const int grid    = (n_edges + EPB - 1) / EPB;

    edge_decoder_wmma<<<grid, 256, 0, stream>>>(
        z_sotu, z_taxon, row, col, w1, b1, w2, b2, out, n_edges);
}
